// GraphAttention_3186865734462
// MI455X (gfx1250) — compile-verified, matrix operand order checked by reading, not run on hardware
//
#include <hip/hip_runtime.h>
#include <hip/hip_bf16.h>
#include <math.h>

// ---------------------------------------------------------------------------
// Graph attention (Graph Transformer conv) for MI455X / gfx1250.
// Dense projections use v_wmma_f32_16x16x32_bf16; sparse passes are
// gather/scatter-atomic and are the HBM-roofline-bound part (~1.6GB traffic).
// ---------------------------------------------------------------------------

#define IN_DIM   128
#define OUT_DIM  128
#define ED_DIM   32
#define NHEAD    8
#define DHEAD    16

typedef __attribute__((ext_vector_type(16))) __bf16 v16bf;
typedef __attribute__((ext_vector_type(8)))  float  v8f;

// ---- ordered-uint encoding for float atomicMax --------------------------
__device__ __forceinline__ unsigned enc_f32(float f) {
    unsigned b = __float_as_uint(f);
    return (b & 0x80000000u) ? ~b : (b | 0x80000000u);
}
__device__ __forceinline__ float dec_f32(unsigned k) {
    unsigned b = (k & 0x80000000u) ? (k & 0x7FFFFFFFu) : ~k;
    return __uint_as_float(b);
}
#define ENC_NEG_INF 0x007FFFFFu   // enc(-inf)

// ---------------------------------------------------------------------------
// K0: init accumulators
// ---------------------------------------------------------------------------
__global__ void init_kernel(float* __restrict__ out_acc, float* __restrict__ denom,
                            unsigned* __restrict__ menc, int N) {
    int i = blockIdx.x * blockDim.x + threadIdx.x;
    if (i < N * OUT_DIM) out_acc[i] = 0.0f;
    if (i < N * NHEAD) { denom[i] = 0.0f; menc[i] = ENC_NEG_INF; }
}

// ---------------------------------------------------------------------------
// K1/K7: Y[n_rows x 128] = X[n_rows x 128] @ W[128 x 128] + bias, bf16 WMMA.
// Block = 256 threads (8 waves), each wave produces a 16x128 strip.
// W is transposed + converted to bf16 in LDS once per block so B-fragment
// loads are contiguous 16B ds reads.
// ---------------------------------------------------------------------------
__global__ __launch_bounds__(256)
void gemm128_bias_wmma(const float* __restrict__ X, const float* __restrict__ W,
                       const float* __restrict__ bias, float* __restrict__ Y,
                       int n_rows) {
    __shared__ __align__(16) __bf16 Wt[128 * 128];   // Wt[n*128+k] = W[k*128+n], 32KB

    const int t = threadIdx.x;
    {   // stage W transposed (each thread: one half-row of W, 64 elements)
        const int k  = t >> 1;
        const int n0 = (t & 1) * 64;
        const float* wrow = W + k * 128 + n0;
        #pragma unroll
        for (int i = 0; i < 64; i += 4) {
            float4 w4 = *(const float4*)(wrow + i);
            const int n = n0 + i;
            Wt[(n + 0) * 128 + k] = (__bf16)w4.x;
            Wt[(n + 1) * 128 + k] = (__bf16)w4.y;
            Wt[(n + 2) * 128 + k] = (__bf16)w4.z;
            Wt[(n + 3) * 128 + k] = (__bf16)w4.w;
        }
    }
    __syncthreads();

    const int wave = t >> 5;
    const int lane = t & 31;
    const int row0 = (blockIdx.x * 8 + wave) * 16;
    if (row0 >= n_rows) return;

    const bool hi  = lane >= 16;
    const int  l16 = lane & 15;
    int arow = row0 + l16;
    if (arow >= n_rows) arow = n_rows - 1;   // clamp (harmless duplicate row)

    union FragBf { v16bf v; uint4 q[2]; };

    v8f acc[8];
    #pragma unroll
    for (int nt = 0; nt < 8; ++nt) acc[nt] = (v8f){0, 0, 0, 0, 0, 0, 0, 0};

    #pragma unroll
    for (int ks = 0; ks < 4; ++ks) {
        const int kbase = ks * 32 + (hi ? 8 : 0);
        // A fragment (ISA 16-bit A 16x32 layout): this lane holds row `arow`,
        // K = kbase..kbase+7 and kbase+16..kbase+23.
        FragBf a;
        {
            const float* xr = X + (size_t)arow * 128 + kbase;
            float4 s0 = *(const float4*)(xr);
            float4 s1 = *(const float4*)(xr + 4);
            float4 s2 = *(const float4*)(xr + 16);
            float4 s3 = *(const float4*)(xr + 20);
            a.v[0]  = (__bf16)s0.x; a.v[1]  = (__bf16)s0.y; a.v[2]  = (__bf16)s0.z; a.v[3]  = (__bf16)s0.w;
            a.v[4]  = (__bf16)s1.x; a.v[5]  = (__bf16)s1.y; a.v[6]  = (__bf16)s1.z; a.v[7]  = (__bf16)s1.w;
            a.v[8]  = (__bf16)s2.x; a.v[9]  = (__bf16)s2.y; a.v[10] = (__bf16)s2.z; a.v[11] = (__bf16)s2.w;
            a.v[12] = (__bf16)s3.x; a.v[13] = (__bf16)s3.y; a.v[14] = (__bf16)s3.z; a.v[15] = (__bf16)s3.w;
        }
        #pragma unroll
        for (int nt = 0; nt < 8; ++nt) {
            const int n = nt * 16 + l16;     // B column owned by this lane
            FragBf b;
            b.q[0] = *(const uint4*)&Wt[n * 128 + kbase];
            b.q[1] = *(const uint4*)&Wt[n * 128 + kbase + 16];
            acc[nt] = __builtin_amdgcn_wmma_f32_16x16x32_bf16(
                false, a.v, false, b.v, (short)0, acc[nt], false, false);
        }
    }

    // Epilogue: C layout — VGPR r holds row (r or r+8), column n0+l16.
    #pragma unroll
    for (int nt = 0; nt < 8; ++nt) {
        const int n  = nt * 16 + l16;
        const float bn = bias[n];
        #pragma unroll
        for (int r = 0; r < 8; ++r) {
            const int m = row0 + r + (hi ? 8 : 0);
            if (m < n_rows) Y[(size_t)m * 128 + n] = acc[nt][r] + bn;
        }
    }
}

// ---------------------------------------------------------------------------
// K2: per-edge bias:  scores[e,h] = edge_attr[e] @ We[:,h] + be[h]
// ---------------------------------------------------------------------------
__global__ __launch_bounds__(256)
void ebias_kernel(const float* __restrict__ edge_attr, const float* __restrict__ We,
                  const float* __restrict__ be, float* __restrict__ scores, int E) {
    __shared__ float sWe[ED_DIM * NHEAD];
    __shared__ float sbe[NHEAD];
    const int t = threadIdx.x;
    if (t < ED_DIM * NHEAD) sWe[t] = We[t];
    if (t < NHEAD) sbe[t] = be[t];
    __syncthreads();

    const int e = blockIdx.x * 256 + t;
    if (e >= E) return;

    const float* ea = edge_attr + (size_t)e * ED_DIM;
    float4 a[8];
    #pragma unroll
    for (int i = 0; i < 8; ++i) a[i] = *(const float4*)(ea + i * 4);

    #pragma unroll
    for (int h = 0; h < NHEAD; ++h) {
        float s = sbe[h];
        #pragma unroll
        for (int i = 0; i < 8; ++i) {
            const int kk = i * 4;
            s += a[i].x * sWe[(kk + 0) * NHEAD + h];
            s += a[i].y * sWe[(kk + 1) * NHEAD + h];
            s += a[i].z * sWe[(kk + 2) * NHEAD + h];
            s += a[i].w * sWe[(kk + 3) * NHEAD + h];
        }
        scores[(size_t)e * NHEAD + h] = s;
    }
}

// ---------------------------------------------------------------------------
// K3: scores[e,h] = 0.25 * <q[dst,h,:], k[src,h,:]> + scores[e,h]; segment max
// ---------------------------------------------------------------------------
__global__ __launch_bounds__(256)
void score_kernel(const float* __restrict__ q, const float* __restrict__ k,
                  const int* __restrict__ ei, float* __restrict__ scores,
                  unsigned* __restrict__ menc, int E) {
    const int t = blockIdx.x * 256 + threadIdx.x;
    if (t >= E * NHEAD) return;
    const int e = t >> 3, h = t & 7;
    const int src = ei[e], dst = ei[E + e];

    const float* qr = q + (size_t)dst * OUT_DIM + h * DHEAD;
    const float* kr = k + (size_t)src * OUT_DIM + h * DHEAD;
    float s = 0.0f;
    #pragma unroll
    for (int i = 0; i < DHEAD; i += 4) {
        float4 qa = *(const float4*)(qr + i);
        float4 ka = *(const float4*)(kr + i);
        s += qa.x * ka.x + qa.y * ka.y + qa.z * ka.z + qa.w * ka.w;
    }
    s = s * 0.25f + scores[t];   // scale = DH^-0.5 = 1/4
    scores[t] = s;
    atomicMax(&menc[(size_t)dst * NHEAD + h], enc_f32(s));
}

// ---------------------------------------------------------------------------
// K4: decode segment max in place; non-finite (empty segment) -> 0
// ---------------------------------------------------------------------------
__global__ void mfix_kernel(unsigned* __restrict__ menc, int total) {
    const int i = blockIdx.x * blockDim.x + threadIdx.x;
    if (i >= total) return;
    float f = dec_f32(menc[i]);
    if (!isfinite(f)) f = 0.0f;
    ((float*)menc)[i] = f;
}

// ---------------------------------------------------------------------------
// K5: e = exp(s - m[dst]); denominator scatter-add
// ---------------------------------------------------------------------------
__global__ __launch_bounds__(256)
void exp_kernel(float* __restrict__ scores, const float* __restrict__ m,
                const int* __restrict__ ei, float* __restrict__ denom, int E) {
    const int t = blockIdx.x * 256 + threadIdx.x;
    if (t >= E * NHEAD) return;
    const int e = t >> 3, h = t & 7;
    const int dst = ei[E + e];
    const float ev = __expf(scores[t] - m[(size_t)dst * NHEAD + h]);
    scores[t] = ev;
    atomicAdd(&denom[(size_t)dst * NHEAD + h], ev);
}

// ---------------------------------------------------------------------------
// K6: out_acc[dst] += v[src] * alpha  (alpha = e / denom[dst])
// ---------------------------------------------------------------------------
__global__ __launch_bounds__(256)
void agg_kernel(const float* __restrict__ scores, const float* __restrict__ denom,
                const float* __restrict__ v, const int* __restrict__ ei,
                float* __restrict__ out_acc, int E) {
    const int t = blockIdx.x * 256 + threadIdx.x;
    if (t >= E * NHEAD) return;
    const int e = t >> 3, h = t & 7;
    const int src = ei[e], dst = ei[E + e];
    const float alpha = scores[t] / denom[(size_t)dst * NHEAD + h];
    const float* vr = v + (size_t)src * OUT_DIM + h * DHEAD;
    float* orow = out_acc + (size_t)dst * OUT_DIM + h * DHEAD;
    #pragma unroll
    for (int i = 0; i < DHEAD; i += 4) {
        float4 va = *(const float4*)(vr + i);
        atomicAdd(&orow[i + 0], va.x * alpha);
        atomicAdd(&orow[i + 1], va.y * alpha);
        atomicAdd(&orow[i + 2], va.z * alpha);
        atomicAdd(&orow[i + 3], va.w * alpha);
    }
}

// ---------------------------------------------------------------------------
// K8: layer norm, one wave32 per row (shfl tree reductions, width 32)
// ---------------------------------------------------------------------------
__global__ __launch_bounds__(256)
void ln_kernel(const float* __restrict__ X, const float* __restrict__ gamma,
               const float* __restrict__ beta, float* __restrict__ Y, int n) {
    const int row  = blockIdx.x * 8 + (threadIdx.x >> 5);
    const int lane = threadIdx.x & 31;
    if (row >= n) return;

    const float* xr = X + (size_t)row * 128;
    float4 vv = *(const float4*)(xr + lane * 4);

    float s = vv.x + vv.y + vv.z + vv.w;
    #pragma unroll
    for (int msk = 16; msk >= 1; msk >>= 1) s += __shfl_xor(s, msk, 32);
    const float mu = s * (1.0f / 128.0f);

    float4 d = {vv.x - mu, vv.y - mu, vv.z - mu, vv.w - mu};
    float ss = d.x * d.x + d.y * d.y + d.z * d.z + d.w * d.w;
    #pragma unroll
    for (int msk = 16; msk >= 1; msk >>= 1) ss += __shfl_xor(ss, msk, 32);
    const float inv = rsqrtf(ss * (1.0f / 128.0f) + 1e-5f);

    const int c = lane * 4;
    float4 g = *(const float4*)(gamma + c);
    float4 b = *(const float4*)(beta + c);
    float4 o = {d.x * inv * g.x + b.x, d.y * inv * g.y + b.y,
                d.z * inv * g.z + b.z, d.w * inv * g.w + b.w};
    *(float4*)(Y + (size_t)row * 128 + c) = o;
}

// ---------------------------------------------------------------------------
extern "C" void kernel_launch(void* const* d_in, const int* in_sizes, int n_in,
                              void* d_out, int out_size, void* d_ws, size_t ws_size,
                              hipStream_t stream) {
    const float* x         = (const float*)d_in[0];
    const float* edge_attr = (const float*)d_in[1];
    const float* Wq = (const float*)d_in[2];
    const float* bq = (const float*)d_in[3];
    const float* Wk = (const float*)d_in[4];
    const float* bk = (const float*)d_in[5];
    const float* Wv = (const float*)d_in[6];
    const float* bv = (const float*)d_in[7];
    const float* We = (const float*)d_in[8];
    const float* be = (const float*)d_in[9];
    const float* Wo = (const float*)d_in[10];
    const float* bo = (const float*)d_in[11];
    const float* gamma = (const float*)d_in[12];
    const float* beta  = (const float*)d_in[13];
    const int*   ei    = (const int*)d_in[14];

    const int N = in_sizes[0] / IN_DIM;
    const int E = in_sizes[14] / 2;

    // workspace layout (floats)
    float* ws      = (float*)d_ws;
    float* q       = ws;
    float* k       = q      + (size_t)N * OUT_DIM;
    float* v       = k      + (size_t)N * OUT_DIM;
    float* scores  = v      + (size_t)N * OUT_DIM;           // E*8
    float* mbuf    = scores + (size_t)E * NHEAD;             // N*8 (unsigned then float)
    float* denom   = mbuf   + (size_t)N * NHEAD;             // N*8
    float* out_acc = denom  + (size_t)N * NHEAD;             // N*128
    float* out2    = q;                                      // reuse q after K3

    const int threads = 256;
    const int gInit  = (N * OUT_DIM + threads - 1) / threads;
    const int gGemm  = (N + 127) / 128;
    const int gEdge  = (E + threads - 1) / threads;
    const int gEH    = (E * NHEAD + threads - 1) / threads;
    const int gM     = (N * NHEAD + threads - 1) / threads;
    const int gLN    = (N + 7) / 8;

    init_kernel<<<gInit, threads, 0, stream>>>(out_acc, denom, (unsigned*)mbuf, N);

    gemm128_bias_wmma<<<gGemm, threads, 0, stream>>>(x, Wq, bq, q, N);
    gemm128_bias_wmma<<<gGemm, threads, 0, stream>>>(x, Wk, bk, k, N);
    gemm128_bias_wmma<<<gGemm, threads, 0, stream>>>(x, Wv, bv, v, N);

    ebias_kernel<<<gEdge, threads, 0, stream>>>(edge_attr, We, be, scores, E);
    score_kernel<<<gEH, threads, 0, stream>>>(q, k, ei, scores, (unsigned*)mbuf, E);
    mfix_kernel<<<gM, threads, 0, stream>>>((unsigned*)mbuf, N * NHEAD);
    exp_kernel<<<gEH, threads, 0, stream>>>(scores, mbuf, ei, denom, E);
    agg_kernel<<<gEH, threads, 0, stream>>>(scores, denom, v, ei, out_acc, E);

    gemm128_bias_wmma<<<gGemm, threads, 0, stream>>>(out_acc, Wo, bo, out2, N);
    ln_kernel<<<gLN, threads, 0, stream>>>(out2, gamma, beta, (float*)d_out, N);
}
